// MoETransformer_70549132804626
// MI455X (gfx1250) — matrix operand order repaired and checked by reading, not compiled, not run on hardware
//
#include <hip/hip_runtime.h>

// ---------------- problem constants ----------------
#define BB 2
#define SS 4096
#define DD 1024
#define HH 16
#define BSZ 64
#define EE 8
#define HD 64
#define NBLK 64
#define MTOT (BB * SS)   // 8192

typedef __bf16 bf16;
typedef __attribute__((ext_vector_type(8)))  __bf16 v8bf;
typedef __attribute__((ext_vector_type(16))) __bf16 v16bf;
typedef __attribute__((ext_vector_type(8)))  float  v8f;

__device__ __forceinline__ v16bf cat8(v8bf lo, v8bf hi) {
  return __builtin_shufflevector(lo, hi, 0,1,2,3,4,5,6,7,8,9,10,11,12,13,14,15);
}

#define WMMA_BF16(a, b, c) \
  __builtin_amdgcn_wmma_f32_16x16x32_bf16(false, (a), false, (b), (short)0, (c), false, false)

// CDNA5 async global->LDS copy (16B per lane), tracked by ASYNCcnt.
#define ASYNC_LD16(ldsoff, gp) \
  asm volatile("global_load_async_to_lds_b128 %0, %1, off" :: "v"(ldsoff), "v"(gp) : "memory")
#define WAIT_ASYNC(n) asm volatile("s_wait_asynccnt " #n ::: "memory")

// ---------------- fp32 -> bf16 convert ----------------
__global__ void k_cvt(const float* __restrict__ src, bf16* __restrict__ dst, int n) {
  int i = (blockIdx.x * blockDim.x + threadIdx.x) * 4;
  if (i + 3 < n) {
    float4 v = *(const float4*)(src + i);
    dst[i + 0] = (bf16)v.x;
    dst[i + 1] = (bf16)v.y;
    dst[i + 2] = (bf16)v.z;
    dst[i + 3] = (bf16)v.w;
  }
}

// convert + transpose a 1024x1024 fp32 matrix [K][N] into bf16 [N][K]
__global__ void k_cvt_t(const float* __restrict__ src, bf16* __restrict__ dst) {
  __shared__ float tile[32][33];
  int bx = blockIdx.x * 32;  // N offset
  int by = blockIdx.y * 32;  // K offset
  int tx = threadIdx.x & 31;
  int ty = threadIdx.x >> 5;  // 0..7
#pragma unroll
  for (int i = 0; i < 4; i++) {
    int k = by + ty + i * 8;
    tile[ty + i * 8][tx] = src[(size_t)k * DD + bx + tx];
  }
  __syncthreads();
#pragma unroll
  for (int i = 0; i < 4; i++) {
    int n = bx + ty + i * 8;
    dst[(size_t)n * DD + by + tx] = (bf16)tile[tx][ty + i * 8];
  }
}

// ---------------- WMMA wave core: 64(M) x 32(N) per wave ----------------
// As: [128 rows][32 k] bf16, Bs: [128 cols][32 k] bf16 (W^T rows)
// 8 waves: wm = w&1 (2 x 64 M), wn = w>>1 (4 x 32 N) -> block tile 128 x 128
// C = 8 frags (64 VGPRs): fits spill-free alongside 4 A + 2 B frags.
__device__ __forceinline__ void mma_step(const bf16* As, const bf16* Bs,
                                         v8f c[4][2], int wm, int wn, int lane) {
  int m  = lane & 15;
  int kg = lane >> 4;
  v16bf a[4];
#pragma unroll
  for (int ms = 0; ms < 4; ms++) {
    const bf16* ap = As + (wm * 64 + ms * 16 + m) * 32;
    a[ms] = *(const v16bf*)(ap + kg * 16);
  }
#pragma unroll
  for (int ns = 0; ns < 2; ns++) {
    const bf16* bp = Bs + (wn * 32 + ns * 16 + m) * 32 + kg * 16;
    v16bf b = *(const v16bf*)bp;
#pragma unroll
    for (int ms = 0; ms < 4; ms++)
      c[ms][ns] = WMMA_BF16(a[ms], b, c[ms][ns]);
  }
}

// ---------------- fused QKV projection GEMM ----------------
// grid (8 Ntiles, 64 Mtiles, 3), block 256
__global__ void k_gemm_qkv(const bf16* __restrict__ xb,
                           const bf16* __restrict__ wqT, const bf16* __restrict__ wkT,
                           const bf16* __restrict__ wvT,
                           const float* __restrict__ bq, const float* __restrict__ bk,
                           const float* __restrict__ bv,
                           bf16* __restrict__ q, bf16* __restrict__ k, bf16* __restrict__ v) {
  const bf16* Wt; const float* bias; bf16* dst;
  if (blockIdx.z == 0)      { Wt = wqT; bias = bq; dst = q; }
  else if (blockIdx.z == 1) { Wt = wkT; bias = bk; dst = k; }
  else                      { Wt = wvT; bias = bv; dst = v; }

  __shared__ __attribute__((aligned(64))) bf16 As[2][128 * 32];
  __shared__ __attribute__((aligned(64))) bf16 Bs[2][128 * 32];
  int tid = threadIdx.x, w = tid >> 5, lane = tid & 31;
  int wm = w & 1, wn = w >> 1;
  int row0 = blockIdx.y * 128, col0 = blockIdx.x * 128;
  int r = tid >> 1, seg2 = (tid & 1) * 2;  // each thread: 2 segs of one row

  v8f c[4][2];
#pragma unroll
  for (int i = 0; i < 4; i++)
#pragma unroll
    for (int j = 0; j < 2; j++) c[i][j] = (v8f)0.0f;

  auto issue = [&](int buf, int kk) {
#pragma unroll
    for (int i = 0; i < 2; i++) {
      int seg = seg2 + i;
      unsigned la = (unsigned)(size_t)&As[buf][r * 32 + seg * 8];
      ASYNC_LD16(la, xb + (size_t)(row0 + r) * DD + kk + seg * 8);
      unsigned lb = (unsigned)(size_t)&Bs[buf][r * 32 + seg * 8];
      ASYNC_LD16(lb, Wt + (size_t)(col0 + r) * DD + kk + seg * 8);
    }
  };

  issue(0, 0);
  for (int i = 0; i < 32; i++) {
    if (i + 1 < 32) {
      issue((i + 1) & 1, (i + 1) * 32);  // next tile in flight while we wait on current
      WAIT_ASYNC(0x4);                   // 4 newest (next tile) may remain outstanding
    } else {
      WAIT_ASYNC(0x0);
    }
    __syncthreads();
    mma_step(&As[i & 1][0], &Bs[i & 1][0], c, wm, wn, lane);
    __syncthreads();
  }

  int hi = lane >> 4, lo = lane & 15;
#pragma unroll
  for (int ms = 0; ms < 4; ms++)
#pragma unroll
    for (int ns = 0; ns < 2; ns++)
#pragma unroll
      for (int rr = 0; rr < 8; rr++) {
        int row = row0 + wm * 64 + ms * 16 + hi * 8 + rr;
        int col = col0 + wn * 32 + ns * 16 + lo;
        dst[(size_t)row * DD + col] = (bf16)(c[ms][ns][rr] + bias[col]);
      }
}

// ---------------- block-local attention ----------------
// grid (H=16, NB=64, B=2), block 128 (4 waves), wave w handles q rows [16w,16w+16)
__global__ void k_attn(const bf16* __restrict__ qb, const bf16* __restrict__ kb,
                       const bf16* __restrict__ vb, bf16* __restrict__ attnb) {
  int h = blockIdx.x, nb = blockIdx.y, b = blockIdx.z;
  int tid = threadIdx.x, w = tid >> 5, lane = tid & 31;
  size_t base = ((size_t)(b * SS + nb * BSZ)) * DD + h * HD;

  __shared__ __attribute__((aligned(64))) bf16 Vt[64 * 64];      // [hd][key]
  __shared__ __attribute__((aligned(64))) bf16 Ps[4][16 * 64];   // per-wave P tile

#pragma unroll
  for (int i = 0; i < 32; i++) {
    int idx = i * 128 + tid;
    int key = idx >> 6, hd = idx & 63;
    Vt[hd * 64 + key] = vb[base + (size_t)key * DD + hd];
  }
  __syncthreads();

  int m = lane & 15, kg = lane >> 4, hi = lane >> 4, lo = lane & 15;
  int qrow = w * 16 + m;

  v8f sc[4];
#pragma unroll
  for (int i = 0; i < 4; i++) sc[i] = (v8f)0.0f;
#pragma unroll
  for (int kk = 0; kk < 2; kk++) {
    const bf16* qp = qb + base + (size_t)qrow * DD + kk * 32;
    v16bf a = *(const v16bf*)(qp + kg * 16);
#pragma unroll
    for (int ns = 0; ns < 4; ns++) {
      int key = ns * 16 + m;
      const bf16* kp = kb + base + (size_t)key * DD + kk * 32 + kg * 16;
      v16bf bfrag = *(const v16bf*)kp;
      sc[ns] = WMMA_BF16(a, bfrag, sc[ns]);
    }
  }

  const float scale = 0.125f;  // 1/sqrt(64)
#pragma unroll
  for (int ns = 0; ns < 4; ns++)
#pragma unroll
    for (int rr = 0; rr < 8; rr++) sc[ns][rr] *= scale;
#pragma unroll
  for (int rr = 0; rr < 8; rr++) {
    float mx = -3.0e38f;
#pragma unroll
    for (int ns = 0; ns < 4; ns++) mx = fmaxf(mx, sc[ns][rr]);
#pragma unroll
    for (int off = 1; off < 16; off <<= 1) mx = fmaxf(mx, __shfl_xor(mx, off, 32));
    float sum = 0.f;
#pragma unroll
    for (int ns = 0; ns < 4; ns++) {
      float e = __expf(sc[ns][rr] - mx);
      sc[ns][rr] = e;
      sum += e;
    }
#pragma unroll
    for (int off = 1; off < 16; off <<= 1) sum += __shfl_xor(sum, off, 32);
    float inv = 1.f / sum;
    int prow = hi * 8 + rr;
#pragma unroll
    for (int ns = 0; ns < 4; ns++)
      Ps[w][prow * 64 + ns * 16 + lo] = (bf16)(sc[ns][rr] * inv);
  }
  __syncthreads();

  v8f oc[4];
#pragma unroll
  for (int i = 0; i < 4; i++) oc[i] = (v8f)0.0f;
#pragma unroll
  for (int kk = 0; kk < 2; kk++) {
    const bf16* pp = &Ps[w][m * 64 + kk * 32];
    v16bf a = *(const v16bf*)(pp + kg * 16);
#pragma unroll
    for (int ns = 0; ns < 4; ns++) {
      const bf16* vp = &Vt[(ns * 16 + m) * 64 + kk * 32 + kg * 16];
      v16bf bfrag = *(const v16bf*)vp;
      oc[ns] = WMMA_BF16(a, bfrag, oc[ns]);
    }
  }
#pragma unroll
  for (int ns = 0; ns < 4; ns++)
#pragma unroll
    for (int rr = 0; rr < 8; rr++) {
      int row = w * 16 + hi * 8 + rr;
      int col = ns * 16 + lo;
      attnb[base + (size_t)row * DD + col] = (bf16)(oc[ns][rr]);
    }
}

// ---------------- output projection + residual ----------------
// grid (8, 64), block 256
__global__ void k_gemm_out(const bf16* __restrict__ ab, const bf16* __restrict__ woT,
                           const float* __restrict__ bo, const float* __restrict__ x,
                           float* __restrict__ hpre) {
  __shared__ __attribute__((aligned(64))) bf16 As[2][128 * 32];
  __shared__ __attribute__((aligned(64))) bf16 Bs[2][128 * 32];
  int tid = threadIdx.x, w = tid >> 5, lane = tid & 31;
  int wm = w & 1, wn = w >> 1;
  int row0 = blockIdx.y * 128, col0 = blockIdx.x * 128;
  int r = tid >> 1, seg2 = (tid & 1) * 2;

  v8f c[4][2];
#pragma unroll
  for (int i = 0; i < 4; i++)
#pragma unroll
    for (int j = 0; j < 2; j++) c[i][j] = (v8f)0.0f;

  auto issue = [&](int buf, int kk) {
#pragma unroll
    for (int i = 0; i < 2; i++) {
      int seg = seg2 + i;
      unsigned la = (unsigned)(size_t)&As[buf][r * 32 + seg * 8];
      ASYNC_LD16(la, ab + (size_t)(row0 + r) * DD + kk + seg * 8);
      unsigned lb = (unsigned)(size_t)&Bs[buf][r * 32 + seg * 8];
      ASYNC_LD16(lb, woT + (size_t)(col0 + r) * DD + kk + seg * 8);
    }
  };

  issue(0, 0);
  for (int i = 0; i < 32; i++) {
    if (i + 1 < 32) {
      issue((i + 1) & 1, (i + 1) * 32);
      WAIT_ASYNC(0x4);
    } else {
      WAIT_ASYNC(0x0);
    }
    __syncthreads();
    mma_step(&As[i & 1][0], &Bs[i & 1][0], c, wm, wn, lane);
    __syncthreads();
  }

  int hi = lane >> 4, lo = lane & 15;
#pragma unroll
  for (int ms = 0; ms < 4; ms++)
#pragma unroll
    for (int ns = 0; ns < 2; ns++)
#pragma unroll
      for (int rr = 0; rr < 8; rr++) {
        int row = row0 + wm * 64 + ms * 16 + hi * 8 + rr;
        int col = col0 + wn * 32 + ns * 16 + lo;
        size_t o = (size_t)row * DD + col;
        hpre[o] = c[ms][ns][rr] + bo[col] + x[o];
      }
}

// ---------------- layernorm -> bf16 ----------------
__global__ void k_ln(const float* __restrict__ hpre, const float* __restrict__ g,
                     const float* __restrict__ bta, bf16* __restrict__ hb) {
  __shared__ float r1[256], r2[256];
  int row = blockIdx.x, tid = threadIdx.x;
  const float* hp = hpre + (size_t)row * DD;
  float v[4], s = 0.f, ss = 0.f;
#pragma unroll
  for (int i = 0; i < 4; i++) {
    v[i] = hp[tid + i * 256];
    s += v[i];
    ss += v[i] * v[i];
  }
  r1[tid] = s; r2[tid] = ss;
  __syncthreads();
  for (int st = 128; st > 0; st >>= 1) {
    if (tid < st) { r1[tid] += r1[tid + st]; r2[tid] += r2[tid + st]; }
    __syncthreads();
  }
  float mean = r1[0] * (1.f / DD);
  float var  = r2[0] * (1.f / DD) - mean * mean;
  float inv  = rsqrtf(var + 1e-5f);
  bf16* hrow = hb + (size_t)row * DD;
#pragma unroll
  for (int i = 0; i < 4; i++) {
    int col = tid + i * 256;
    hrow[col] = (bf16)((v[i] - mean) * inv * g[col] + bta[col]);
  }
}

// ---------------- MoE gating: top-1 + counts ----------------
__global__ void k_gate(const bf16* __restrict__ hb, const float* __restrict__ Wg,
                       int* __restrict__ eidx, float* __restrict__ gatew, int* __restrict__ ctrl) {
  int token = blockIdx.x * 8 + (threadIdx.x >> 5);
  int lane = threadIdx.x & 31;
  const bf16* hrow = hb + (size_t)token * DD;
  float acc[EE];
#pragma unroll
  for (int e = 0; e < EE; e++) acc[e] = 0.f;
  for (int j = 0; j < 32; j++) {
    int d0 = j * 32 + lane;
    float hv = (float)hrow[d0];
#pragma unroll
    for (int e = 0; e < EE; e++) acc[e] += hv * Wg[d0 * EE + e];
  }
#pragma unroll
  for (int e = 0; e < EE; e++)
#pragma unroll
    for (int off = 16; off > 0; off >>= 1) acc[e] += __shfl_xor(acc[e], off, 32);
  if (lane == 0) {
    int best = 0; float bm = acc[0];
#pragma unroll
    for (int e = 1; e < EE; e++)
      if (acc[e] > bm) { bm = acc[e]; best = e; }
    float sum = 0.f;
#pragma unroll
    for (int e = 0; e < EE; e++) sum += __expf(acc[e] - bm);
    eidx[token] = best;
    gatew[token] = 1.f / sum;
    atomicAdd(&ctrl[best], 1);
  }
}

__global__ void k_offsets(int* ctrl) {
  int s = 0;
  for (int e = 0; e < EE; e++) { ctrl[8 + e] = s; s += ctrl[e]; }
}

__global__ void k_scatter(const int* __restrict__ eidx, int* __restrict__ ctrl,
                          int* __restrict__ tokens) {
  int t = blockIdx.x * 256 + threadIdx.x;
  int e = eidx[t];
  int pos = atomicAdd(&ctrl[16 + e], 1);
  tokens[ctrl[8 + e] + pos] = t;
}

// ---------------- routed expert GEMM -> final output ----------------
// grid (8 Ntiles, 64 Mtiles, 8 experts), block 256
__global__ void k_expert(const bf16* __restrict__ hb, const bf16* __restrict__ weT,
                         const float* __restrict__ be, const int* __restrict__ tokens,
                         const int* __restrict__ ctrl, const float* __restrict__ gatew,
                         float* __restrict__ out) {
  int e = blockIdx.z;
  int cnt = ctrl[e];
  int m0 = blockIdx.y * 128;
  if (m0 >= cnt) return;
  int off = ctrl[8 + e];

  __shared__ int toks[128];
  __shared__ __attribute__((aligned(64))) bf16 As[2][128 * 32];
  __shared__ __attribute__((aligned(64))) bf16 Bs[2][128 * 32];
  int tid = threadIdx.x;
  if (tid < 128) {
    int i = m0 + tid;
    toks[tid] = (i < cnt) ? tokens[off + i] : -1;
  }
  __syncthreads();

  const bf16* Wt = weT + (size_t)e * DD * DD;
  int n0 = blockIdx.x * 128;
  int w = tid >> 5, lane = tid & 31, wm = w & 1, wn = w >> 1;
  int r = tid >> 1, seg2 = (tid & 1) * 2;
  int tokr = toks[r];
  const bf16* arow = hb + (size_t)(tokr < 0 ? 0 : tokr) * DD;

  v8f c[4][2];
#pragma unroll
  for (int i = 0; i < 4; i++)
#pragma unroll
    for (int j = 0; j < 2; j++) c[i][j] = (v8f)0.0f;

  auto issue = [&](int buf, int kk) {
#pragma unroll
    for (int i = 0; i < 2; i++) {
      int seg = seg2 + i;
      unsigned la = (unsigned)(size_t)&As[buf][r * 32 + seg * 8];
      ASYNC_LD16(la, arow + kk + seg * 8);
      unsigned lb = (unsigned)(size_t)&Bs[buf][r * 32 + seg * 8];
      ASYNC_LD16(lb, Wt + (size_t)(n0 + r) * DD + kk + seg * 8);
    }
  };

  issue(0, 0);
  for (int i = 0; i < 32; i++) {
    if (i + 1 < 32) {
      issue((i + 1) & 1, (i + 1) * 32);
      WAIT_ASYNC(0x4);
    } else {
      WAIT_ASYNC(0x0);
    }
    __syncthreads();
    mma_step(&As[i & 1][0], &Bs[i & 1][0], c, wm, wn, lane);
    __syncthreads();
  }

  int hi = lane >> 4, lo = lane & 15;
#pragma unroll
  for (int ms = 0; ms < 4; ms++)
#pragma unroll
    for (int ns = 0; ns < 2; ns++)
#pragma unroll
      for (int rr = 0; rr < 8; rr++) {
        int rl = wm * 64 + ms * 16 + hi * 8 + rr;
        int tok = toks[rl];
        if (tok >= 0) {
          int col = n0 + wn * 32 + ns * 16 + lo;
          out[(size_t)tok * DD + col] = gatew[tok] * (c[ms][ns][rr] + be[e * DD + col]);
        }
      }
}

// ---------------- host: workspace layout + launches ----------------
extern "C" void kernel_launch(void* const* d_in, const int* in_sizes, int n_in,
                              void* d_out, int out_size, void* d_ws, size_t ws_size,
                              hipStream_t stream) {
  const float* x  = (const float*)d_in[0];
  const float* Wq = (const float*)d_in[1];
  const float* bq = (const float*)d_in[2];
  const float* Wk = (const float*)d_in[3];
  const float* bk = (const float*)d_in[4];
  const float* Wv = (const float*)d_in[5];
  const float* bv = (const float*)d_in[6];
  const float* Wo = (const float*)d_in[7];
  const float* bo = (const float*)d_in[8];
  const float* gamma = (const float*)d_in[9];
  const float* beta  = (const float*)d_in[10];
  const float* Wg = (const float*)d_in[11];
  const float* We = (const float*)d_in[12];
  const float* be = (const float*)d_in[13];
  float* out = (float*)d_out;

  char* w = (char*)d_ws;
  size_t off = 0;
  auto alloc = [&](size_t bytes) -> char* {
    char* p = w + off;
    off = (off + bytes + 255) & ~(size_t)255;
    return p;
  };
  const size_t XE = (size_t)MTOT * DD;
  bf16* xb    = (bf16*)alloc(XE * 2);
  bf16* qb    = (bf16*)alloc(XE * 2);
  bf16* kb    = (bf16*)alloc(XE * 2);
  bf16* vb    = (bf16*)alloc(XE * 2);
  bf16* wqT   = (bf16*)alloc((size_t)DD * DD * 2);
  bf16* wkT   = (bf16*)alloc((size_t)DD * DD * 2);
  bf16* wvT   = (bf16*)alloc((size_t)DD * DD * 2);
  bf16* woT   = (bf16*)alloc((size_t)DD * DD * 2);
  bf16* weT   = (bf16*)alloc((size_t)EE * DD * DD * 2);
  bf16* attnb = (bf16*)alloc(XE * 2);
  float* hpre = (float*)alloc(XE * 4);
  bf16* hb    = (bf16*)alloc(XE * 2);
  float* gatew = (float*)alloc((size_t)MTOT * 4);
  int* eidx    = (int*)alloc((size_t)MTOT * 4);
  int* tokens  = (int*)alloc((size_t)MTOT * 4);
  int* ctrl    = (int*)alloc(256);  // [0..7] counts, [8..15] offsets, [16..23] cursors
  (void)ws_size; (void)in_sizes; (void)n_in; (void)out_size;

  k_cvt<<<(XE / 4 + 255) / 256, 256, 0, stream>>>(x, xb, (int)XE);
  dim3 tg(32, 32);
  k_cvt_t<<<tg, 256, 0, stream>>>(Wq, wqT);
  k_cvt_t<<<tg, 256, 0, stream>>>(Wk, wkT);
  k_cvt_t<<<tg, 256, 0, stream>>>(Wv, wvT);
  k_cvt_t<<<tg, 256, 0, stream>>>(Wo, woT);
  for (int e = 0; e < EE; e++)
    k_cvt_t<<<tg, 256, 0, stream>>>(We + (size_t)e * DD * DD, weT + (size_t)e * DD * DD);
  hipMemsetAsync(ctrl, 0, 256, stream);

  k_gemm_qkv<<<dim3(8, 64, 3), 256, 0, stream>>>(xb, wqT, wkT, wvT, bq, bk, bv, qb, kb, vb);
  k_attn<<<dim3(HH, NBLK, BB), 128, 0, stream>>>(qb, kb, vb, attnb);
  k_gemm_out<<<dim3(8, 64), 256, 0, stream>>>(attnb, woT, bo, x, hpre);
  k_ln<<<MTOT, 256, 0, stream>>>(hpre, gamma, beta, hb);
  k_gate<<<MTOT / 8, 256, 0, stream>>>(hb, Wg, eidx, gatew, ctrl);
  k_offsets<<<1, 1, 0, stream>>>(ctrl);
  k_scatter<<<MTOT / 256, 256, 0, stream>>>(eidx, ctrl, tokens);
  k_expert<<<dim3(8, 64, 8), 256, 0, stream>>>(hb, weT, be, tokens, ctrl, gatew, out);
}
